// DWAModel_64390149702175
// MI455X (gfx1250) — matrix operand order repaired and checked by reading, not compiled, not run on hardware
//
#include <hip/hip_runtime.h>
#include <hip/hip_bf16.h>
#include <math.h>

// ---- problem constants (match reference) ----
#define BB   4
#define TT   512
#define VV   32000
#define DA   512
#define DB   512
#define DFF  2048
#define NH   8
#define HD   64
#define NPOOL 32768
#define RR   2
#define AA   4
#define DKQ  64
#define KMAXK 16
#define DD   2048   // R*(DA+DB)

typedef __attribute__((ext_vector_type(16))) __bf16 v16bf;
typedef __attribute__((ext_vector_type(8)))  float  v8f;

union Frag16 { v16bf v; unsigned short h[16]; uint4 q[2]; };
union Frag8f { v8f  v; float f[8]; };

__device__ __forceinline__ unsigned short f2bf(float x) {
  union { float f; unsigned u; } c; c.f = x;
  unsigned u = c.u;
  return (unsigned short)((u + 0x7FFFu + ((u >> 16) & 1u)) >> 16);
}

__device__ __forceinline__ float gelu_tanh(float x) {
  return 0.5f * x * (1.f + tanhf(0.7978845608028654f * (x + 0.044715f * x * x * x)));
}

// A-fragment loaders (16x32 bf16 operand, ISA 16-bit A layout)
__device__ __forceinline__ void loadA(Frag16& af, const unsigned short* __restrict__ A,
                                      size_t row, int K, int kk, int alo) {
  const unsigned short* ap = A + row * (size_t)K + kk + alo;
  __builtin_prefetch(ap + 32, 0, 1);
  af.q[0] = *reinterpret_cast<const uint4*>(ap);        // j0..7  -> K = kk+alo..+7
  af.q[1] = *reinterpret_cast<const uint4*>(ap + 16);   // j8..15 -> K = kk+16+alo..
}
__device__ __forceinline__ void loadA(Frag16& af, const float* __restrict__ A,
                                      size_t row, int K, int kk, int alo) {
  const float* ap = A + row * (size_t)K + kk + alo;
  __builtin_prefetch(ap + 32, 0, 1);
  float4 a0 = *reinterpret_cast<const float4*>(ap);
  float4 a1 = *reinterpret_cast<const float4*>(ap + 4);
  float4 a2 = *reinterpret_cast<const float4*>(ap + 16);
  float4 a3 = *reinterpret_cast<const float4*>(ap + 20);
  af.h[0]=f2bf(a0.x); af.h[1]=f2bf(a0.y); af.h[2]=f2bf(a0.z); af.h[3]=f2bf(a0.w);
  af.h[4]=f2bf(a1.x); af.h[5]=f2bf(a1.y); af.h[6]=f2bf(a1.z); af.h[7]=f2bf(a1.w);
  af.h[8]=f2bf(a2.x); af.h[9]=f2bf(a2.y); af.h[10]=f2bf(a2.z); af.h[11]=f2bf(a2.w);
  af.h[12]=f2bf(a3.x); af.h[13]=f2bf(a3.y); af.h[14]=f2bf(a3.z); af.h[15]=f2bf(a3.w);
}

__device__ __forceinline__ void storeC(float* p, float v) { *p = v; }
__device__ __forceinline__ void storeC(unsigned short* p, float v) { *p = f2bf(v); }

// ------------------------------------------------------------------
// WMMA GEMM: C[b] = act(A[b] @ Bt[b]^T + bias) (+ resid)
// A: AT [M,K] row-major (bf16 or f32). Bt: bf16 [N,K] row-major (pre-transposed).
// C: CT (f32 or bf16). M mult 64, N mult 16, K mult 32.
// 4 waves/block; B tile (16 cols x 32 k) DMA'd to LDS with
// GLOBAL_LOAD_ASYNC_TO_LDS_B64 (ASYNCcnt), shared by all 4 waves.
// ------------------------------------------------------------------
template <typename AT, typename CT>
__global__ void __launch_bounds__(128) k_gemm(
    const AT* __restrict__ A, const unsigned short* __restrict__ Bt, CT* __restrict__ C,
    int M, int N, int K, long sA, long sB, long sC,
    const float* __restrict__ bias, const float* __restrict__ resid, int act)
{
  __shared__ unsigned short Blt[16][32];   // [col][k] tile (already transposed in memory)

  const int batch = blockIdx.z;
  A  += (long)batch * sA;
  Bt += (long)batch * sB;
  C  += (long)batch * sC;

  const int tid  = threadIdx.x;
  const int wave = tid >> 5;
  const int lane = tid & 31;
  const int l15  = lane & 15;
  const int hi   = lane >> 4;
  const int alo  = hi * 8;
  const int n0   = blockIdx.x * 16;
  const int m0   = blockIdx.y * 64 + wave * 16;
  const size_t row = (size_t)(m0 + l15);
  const int col = n0 + l15;

  // B staging assignment: thread -> (col-row of Bt, 8-byte part)
  const int nrow = tid >> 3;   // 0..15
  const int part = tid & 7;    // 0..7  (4 ushorts = 8 bytes each)

  Frag8f acc;
#pragma unroll
  for (int i = 0; i < 8; ++i) acc.f[i] = 0.f;

  for (int kk = 0; kk < K; kk += 32) {
    // async DMA: contiguous 8B of Bt[(n0+nrow), kk+part*4 ..] -> LDS Blt[nrow][part*4]
    {
      const unsigned short* gsrc = Bt + (size_t)(n0 + nrow) * K + kk + part * 4;
      unsigned short* ldst = &Blt[nrow][part * 4];
      unsigned long long ga = (unsigned long long)(size_t)gsrc;
      unsigned lo = (unsigned)(size_t)ldst;
      asm volatile("global_load_async_to_lds_b64 %0, %1, off"
                   :: "v"(lo), "v"(ga) : "memory");
    }

    Frag16 af;
    loadA(af, A, row, K, kk, alo);

    asm volatile("s_wait_asynccnt 0" ::: "memory");
    __syncthreads();

    Frag16 bf;
    const unsigned short* bp = &Blt[l15][hi * 8];
    bf.q[0] = *reinterpret_cast<const uint4*>(bp);
    bf.q[1] = *reinterpret_cast<const uint4*>(bp + 16);

    acc.v = __builtin_amdgcn_wmma_f32_16x16x32_bf16(
        false, af.v, false, bf.v, (short)0, acc.v, false, false);
    __syncthreads();
  }

  const float bv = bias ? bias[col] : 0.f;
#pragma unroll
  for (int v = 0; v < 8; ++v) {
    const int r = m0 + v + hi * 8;
    const size_t idx = (size_t)r * N + col;
    float val = acc.f[v] + bv;
    if (act == 1) val = gelu_tanh(val);
    if (resid) val += resid[idx];
    storeC(&C[idx], val);
  }
}

// ------------------------------------------------------------------
// Fused causal attention (flash-style, online softmax), bf16 in/out.
// qkv: bf16 [B,T,3*DA]; out: bf16 [B,T,DA]. grid=(T/64, B*H), 128 thr.
// ------------------------------------------------------------------
__global__ void __launch_bounds__(128) k_attn(const unsigned short* __restrict__ qkv,
                                              unsigned short* __restrict__ out)
{
  __shared__ unsigned short Plds[4][16][32];  // per-wave P tile (bf16)

  const int wave = threadIdx.x >> 5;
  const int lane = threadIdx.x & 31;
  const int l15  = lane & 15;
  const int hi   = lane >> 4;
  const int alo  = hi * 8;
  const int bh = blockIdx.y;
  const int b = bh / NH, h = bh % NH;
  const int q0 = blockIdx.x * 64 + wave * 16;
  const unsigned short* base = qkv + (size_t)b * TT * (3 * DA);

  // Q operand fragments (constant over key loop)
  Frag16 aq[2];
  const int qrowA = q0 + l15;
#pragma unroll
  for (int ks = 0; ks < 2; ++ks) {
    const unsigned short* qp = base + (size_t)qrowA * (3 * DA) + h * HD + ks * 32;
    aq[ks].q[0] = *reinterpret_cast<const uint4*>(qp + alo);
    aq[ks].q[1] = *reinterpret_cast<const uint4*>(qp + 16 + alo);
  }

  Frag8f o[4];
#pragma unroll
  for (int t = 0; t < 4; ++t)
#pragma unroll
    for (int i = 0; i < 8; ++i) o[t].f[i] = 0.f;
  float m_row[8], l_row[8];
#pragma unroll
  for (int i = 0; i < 8; ++i) { m_row[i] = -1e30f; l_row[i] = 0.f; }

  const int kend = blockIdx.x * 64 + 64;   // causal bound (uniform per block)
  for (int kb = 0; kb < kend; kb += 32) {
    // ---- S = Q K^T (two 16-key tiles, K contiguous loads in bf16) ----
    Frag8f s[2];
#pragma unroll
    for (int ss = 0; ss < 2; ++ss) {
#pragma unroll
      for (int i = 0; i < 8; ++i) s[ss].f[i] = 0.f;
      const int keyc = kb + ss * 16 + l15;
#pragma unroll
      for (int ks = 0; ks < 2; ++ks) {
        Frag16 bk;
        const unsigned short* kp =
            base + (size_t)keyc * (3 * DA) + DA + h * HD + ks * 32 + hi * 8;
        bk.q[0] = *reinterpret_cast<const uint4*>(kp);
        bk.q[1] = *reinterpret_cast<const uint4*>(kp + 16);
        s[ss].v = __builtin_amdgcn_wmma_f32_16x16x32_bf16(
            false, aq[ks].v, false, bk.v, (short)0, s[ss].v, false, false);
      }
#pragma unroll
      for (int v = 0; v < 8; ++v) {
        const int qr = q0 + v + hi * 8;
        const float sv = s[ss].f[v] * 0.125f;        // 1/sqrt(64)
        s[ss].f[v] = (keyc <= qr) ? sv : -1e30f;     // causal mask
      }
    }

    // ---- online softmax (half-wave shfl reductions; wave32) ----
#pragma unroll
    for (int v = 0; v < 8; ++v) {
      float mv = fmaxf(s[0].f[v], s[1].f[v]);
#pragma unroll
      for (int off = 1; off < 16; off <<= 1) mv = fmaxf(mv, __shfl_xor(mv, off, 32));
      const float m_new = fmaxf(m_row[v], mv);
      const float scale = __expf(m_row[v] - m_new);
      const float p0 = __expf(s[0].f[v] - m_new);
      const float p1 = __expf(s[1].f[v] - m_new);
      s[0].f[v] = p0; s[1].f[v] = p1;
      float ls = p0 + p1;
#pragma unroll
      for (int off = 1; off < 16; off <<= 1) ls += __shfl_xor(ls, off, 32);
      l_row[v] = l_row[v] * scale + ls;
      m_row[v] = m_new;
#pragma unroll
      for (int t = 0; t < 4; ++t) o[t].f[v] *= scale;
    }

    // ---- re-layout P (16x32) via LDS into A operand ----
#pragma unroll
    for (int v = 0; v < 8; ++v) {
      const int r = v + hi * 8;
      Plds[wave][r][l15]      = f2bf(s[0].f[v]);
      Plds[wave][r][16 + l15] = f2bf(s[1].f[v]);
    }
    __syncthreads();
    Frag16 pa;
    {
      const unsigned short* pp = &Plds[wave][l15][0];
      pa.q[0] = *reinterpret_cast<const uint4*>(pp + alo);
      pa.q[1] = *reinterpret_cast<const uint4*>(pp + 16 + alo);
    }
    __syncthreads();

    // ---- O += P @ V (hd=64 -> 4 n-tiles) ----
#pragma unroll
    for (int t = 0; t < 4; ++t) {
      Frag16 bv;
      const int dcol = t * 16 + l15;
#pragma unroll
      for (int j = 0; j < 16; ++j) {
        const int key = kb + hi * 8 + j + (j >= 8 ? 8 : 0);
        bv.h[j] = base[(size_t)key * (3 * DA) + 2 * DA + h * HD + dcol];
      }
      o[t].v = __builtin_amdgcn_wmma_f32_16x16x32_bf16(
          false, pa.v, false, bv.v, (short)0, o[t].v, false, false);
    }
  }

  // ---- normalize & write bf16 [B,T,DA] ----
#pragma unroll
  for (int t = 0; t < 4; ++t) {
    const int dcol = t * 16 + l15;
#pragma unroll
    for (int v = 0; v < 8; ++v) {
      const int qr = q0 + v + hi * 8;
      out[((size_t)b * TT + qr) * DA + h * HD + dcol] = f2bf(o[t].f[v] / l_row[v]);
    }
  }
}

// ------------------------------------------------------------------
// Helper kernels
// ------------------------------------------------------------------
__global__ void k_cast(const float* __restrict__ s, unsigned short* __restrict__ d, long n)
{
  long i = ((long)blockIdx.x * blockDim.x + threadIdx.x) * 4;
  if (i + 3 < n) {
    float4 v = *reinterpret_cast<const float4*>(s + i);
    d[i] = f2bf(v.x); d[i + 1] = f2bf(v.y); d[i + 2] = f2bf(v.z); d[i + 3] = f2bf(v.w);
  } else {
    for (; i < n; ++i) d[i] = f2bf(s[i]);
  }
}

// transpose-cast: s [rows, cols] f32 -> d [cols, rows] bf16 (coalesced writes)
__global__ void k_cast_t(const float* __restrict__ s, unsigned short* __restrict__ d,
                         int rows, int cols)
{
  const long idx = (long)blockIdx.x * blockDim.x + threadIdx.x;
  if (idx >= (long)rows * cols) return;
  const int r = (int)(idx % rows);
  const int c = (int)(idx / rows);
  d[idx] = f2bf(s[(size_t)r * cols + c]);
}

__global__ void k_embed(const int* __restrict__ ids, const float* __restrict__ emb,
                        float* __restrict__ x)
{
  const int idx = blockIdx.x * blockDim.x + threadIdx.x;
  if (idx >= BB * TT * DA) return;
  const int d = idx % DA;
  const int bt = idx / DA;
  const int t = bt % TT;
  float pe;
  if (d < DA / 2) {
    const float ang = (float)t * __expf(-((2.f * d) / DA) * 9.210340371976184f); // ln(1e4)
    pe = __sinf(ang);
  } else {
    const int i = d - DA / 2;
    const float ang = (float)t * __expf(-((2.f * i) / DA) * 9.210340371976184f);
    pe = __cosf(ang);
  }
  x[idx] = emb[(size_t)ids[bt] * DA + d] + pe;
}

template <typename OT>
__global__ void k_ln(const float* __restrict__ src, OT* __restrict__ dst,
                     const float* __restrict__ g, const float* __restrict__ be, int dim)
{
  __shared__ float r1[256], r2[256];
  const int row = blockIdx.x;
  const float* x = src + (size_t)row * dim;
  float s = 0.f, s2 = 0.f;
  for (int i = threadIdx.x; i < dim; i += blockDim.x) { const float v = x[i]; s += v; s2 += v * v; }
  r1[threadIdx.x] = s; r2[threadIdx.x] = s2;
  __syncthreads();
  for (int off = blockDim.x >> 1; off > 0; off >>= 1) {
    if ((int)threadIdx.x < off) { r1[threadIdx.x] += r1[threadIdx.x + off]; r2[threadIdx.x] += r2[threadIdx.x + off]; }
    __syncthreads();
  }
  const float mean = r1[0] / dim;
  const float var  = r2[0] / dim - mean * mean;
  const float inv  = rsqrtf(var + 1e-5f);
  OT* y = dst + (size_t)row * dim;
  for (int i = threadIdx.x; i < dim; i += blockDim.x) storeC(&y[i], (x[i] - mean) * inv * g[i] + be[i]);
}

__global__ void k_rowmean(const float* __restrict__ x, float* __restrict__ z)
{
  const int idx = blockIdx.x * blockDim.x + threadIdx.x;
  if (idx >= BB * DA) return;
  const int b = idx / DA, d = idx % DA;
  float s = 0.f;
  for (int t = 0; t < TT; ++t) s += x[((size_t)b * TT + t) * DA + d];
  z[idx] = s / (float)TT;
}

__global__ void k_qvec(const float* __restrict__ z, const float* __restrict__ wq,
                       float* __restrict__ qv)
{
  const int idx = blockIdx.x * blockDim.x + threadIdx.x;
  if (idx >= BB * AA * DKQ) return;
  const int k = idx % DKQ;
  const int a = (idx / DKQ) % AA;
  const int b = idx / (DKQ * AA);
  float s = 0.f;
  for (int d = 0; d < DA; ++d) s += z[b * DA + d] * wq[((size_t)a * DA + d) * DKQ + k];
  qv[idx] = s;
}

__global__ void k_comb(const float* __restrict__ qv, const float* __restrict__ keys,
                       float* __restrict__ comb)
{
  const int idx = blockIdx.x * blockDim.x + threadIdx.x;
  if (idx >= BB * NPOOL) return;
  const int n = idx % NPOOL, b = idx / NPOOL;
  float s = 0.f;
  for (int a = 0; a < AA; ++a) {
    const float* qp = qv + ((size_t)b * AA + a) * DKQ;
    const float* kp = keys + ((size_t)a * NPOOL + n) * DKQ;
    for (int k = 0; k < DKQ; ++k) s += qp[k] * kp[k];
  }
  comb[idx] = s * (1.f / 32.f);   // /sqrt(DK)=8, then mean over A=4
}

__global__ void k_topk(float* __restrict__ comb, float* __restrict__ tvals,
                       int* __restrict__ tidx)
{
  __shared__ float sv[256];
  __shared__ int   si[256];
  const int b = blockIdx.x;
  float* c = comb + (size_t)b * NPOOL;
  for (int pass = 0; pass < KMAXK; ++pass) {
    float best = -1e30f; int bi = NPOOL;
    for (int n = threadIdx.x; n < NPOOL; n += 256) {
      const float v = c[n];
      if (v > best) { best = v; bi = n; }
    }
    sv[threadIdx.x] = best; si[threadIdx.x] = bi;
    __syncthreads();
    for (int off = 128; off > 0; off >>= 1) {
      if ((int)threadIdx.x < off) {
        const float v2 = sv[threadIdx.x + off]; const int i2 = si[threadIdx.x + off];
        if (v2 > sv[threadIdx.x] || (v2 == sv[threadIdx.x] && i2 < si[threadIdx.x])) {
          sv[threadIdx.x] = v2; si[threadIdx.x] = i2;
        }
      }
      __syncthreads();
    }
    if (threadIdx.x == 0) {
      tvals[b * KMAXK + pass] = sv[0];
      tidx[b * KMAXK + pass]  = si[0];
      c[si[0]] = -1e30f;
    }
    __syncthreads();
  }
}

__global__ void k_alphas(const float* __restrict__ tvals, const float* __restrict__ lam,
                         const unsigned char* __restrict__ warm, float* __restrict__ alphas)
{ // softmax(top_vals / (warmup ? 1 : lambda)) — renormalized-soft_top identity
  const int b = threadIdx.x;
  if (b >= BB) return;
  const float l = warm[0] ? 1.f : lam[0];
  float mx = -1e30f;
  for (int k = 0; k < KMAXK; ++k) mx = fmaxf(mx, tvals[b * KMAXK + k] / l);
  float e[KMAXK]; float s = 0.f;
  for (int k = 0; k < KMAXK; ++k) { e[k] = __expf(tvals[b * KMAXK + k] / l - mx); s += e[k]; }
  for (int k = 0; k < KMAXK; ++k) alphas[b * KMAXK + k] = e[k] / s;
}

__global__ void k_mixed(const float* __restrict__ alphas, const int* __restrict__ tidx,
                        const float* __restrict__ pool, float* __restrict__ mixed)
{
  const int idx = blockIdx.x * blockDim.x + threadIdx.x;
  if (idx >= BB * DD) return;
  const int d = idx % DD, b = idx / DD;
  float s = 0.f;
  for (int k = 0; k < KMAXK; ++k)
    s += alphas[b * KMAXK + k] * pool[(size_t)tidx[b * KMAXK + k] * DD + d];
  mixed[idx] = s;
}

__global__ void k_wassm(const float* __restrict__ mixed, const float* __restrict__ wbase,
                        const float* __restrict__ gamma, unsigned short* __restrict__ Wt)
{ // Wt[b,o,d] = wbase[d,o] + gamma * sum_r U[b,d,r]*V[b,r,o]   (bf16, transposed)
  const int idx = blockIdx.x * blockDim.x + threadIdx.x;
  if (idx >= BB * DA * DB) return;
  const int d = idx % DA;
  const int o = (idx / DA) % DB;
  const int b = idx / (DA * DB);
  const float* mx = mixed + (size_t)b * DD;
  const float u0 = mx[d * RR + 0], u1 = mx[d * RR + 1];
  const float v0 = mx[DA * RR + 0 * DB + o], v1 = mx[DA * RR + 1 * DB + o];
  Wt[idx] = f2bf(wbase[(size_t)d * DB + o] + gamma[0] * (u0 * v0 + u1 * v1));
}

// ------------------------------------------------------------------
extern "C" void kernel_launch(void* const* d_in, const int* in_sizes, int n_in,
                              void* d_out, int out_size, void* d_ws, size_t ws_size,
                              hipStream_t stream) {
  (void)in_sizes; (void)n_in; (void)out_size; (void)ws_size;
  const int*   ids     = (const int*)d_in[0];
  const float* lam     = (const float*)d_in[1];
  const unsigned char* warm = (const unsigned char*)d_in[2];
  const float* emb     = (const float*)d_in[3];
  const float* pool    = (const float*)d_in[4];
  const float* wq      = (const float*)d_in[5];
  const float* wk      = (const float*)d_in[6];
  const float* w_base  = (const float*)d_in[7];
  const float* b_base  = (const float*)d_in[8];
  const float* gamma   = (const float*)d_in[9];
  const float* mid_g   = (const float*)d_in[10];
  const float* mid_b   = (const float*)d_in[11];
  const float* a_ln1g  = (const float*)d_in[12];
  const float* a_ln1b  = (const float*)d_in[13];
  const float* a_wqkv  = (const float*)d_in[14];
  const float* a_wo    = (const float*)d_in[15];
  const float* a_ln2g  = (const float*)d_in[16];
  const float* a_ln2b  = (const float*)d_in[17];
  const float* a_w1    = (const float*)d_in[18];
  const float* a_w2    = (const float*)d_in[19];
  const float* b_ln1g  = (const float*)d_in[20];
  const float* b_ln1b  = (const float*)d_in[21];
  const float* b_wqkv  = (const float*)d_in[22];
  const float* b_wo    = (const float*)d_in[23];
  const float* b_ln2g  = (const float*)d_in[24];
  const float* b_ln2b  = (const float*)d_in[25];
  const float* b_w1    = (const float*)d_in[26];
  const float* b_w2    = (const float*)d_in[27];
  const float* lm_head = (const float*)d_in[28];
  float* out = (float*)d_out;

  // ---- workspace carve (byte cursor, 256B aligned chunks) ----
  char* wsb = (char*)d_ws;
  size_t cur = 0;
  auto allocB = [&](size_t bytes) { char* p = wsb + cur; cur += (bytes + 255) & ~(size_t)255; return p; };
  // fp32
  float* x     = (float*)allocB((size_t)BB * TT * DA * 4);
  float* hmid  = (float*)allocB((size_t)BB * TT * DB * 4);
  float* keys  = (float*)allocB((size_t)AA * NPOOL * DKQ * 4);
  float* comb  = (float*)allocB((size_t)BB * NPOOL * 4);
  float* z     = (float*)allocB((size_t)BB * DA * 4);
  float* qv    = (float*)allocB((size_t)BB * AA * DKQ * 4);
  float* tvals = (float*)allocB((size_t)BB * KMAXK * 4);
  float* alph  = (float*)allocB((size_t)BB * KMAXK * 4);
  int*   tidx  = (int*)allocB((size_t)BB * KMAXK * 4);
  float* mix   = (float*)allocB((size_t)BB * DD * 4);
  // bf16 activations
  unsigned short* lnb_bf = (unsigned short*)allocB((size_t)BB * TT * DA * 2);
  unsigned short* qkv_bf = (unsigned short*)allocB((size_t)BB * TT * 3 * DA * 2);
  unsigned short* att_bf = (unsigned short*)allocB((size_t)BB * TT * DA * 2);
  unsigned short* ffn_bf = (unsigned short*)allocB((size_t)BB * TT * DFF * 2);
  unsigned short* x_bf   = (unsigned short*)allocB((size_t)BB * TT * DA * 2);
  unsigned short* h_bf   = (unsigned short*)allocB((size_t)BB * TT * DB * 2);
  unsigned short* Wt_bf  = (unsigned short*)allocB((size_t)BB * DA * DB * 2);
  // bf16 transposed weights (converted once per call)
  unsigned short* wqkvA_t = (unsigned short*)allocB((size_t)DA * 3 * DA * 2);
  unsigned short* woA_t   = (unsigned short*)allocB((size_t)DA * DA * 2);
  unsigned short* w1A_t   = (unsigned short*)allocB((size_t)DA * DFF * 2);
  unsigned short* w2A_t   = (unsigned short*)allocB((size_t)DFF * DA * 2);
  unsigned short* wqkvB_t = (unsigned short*)allocB((size_t)DB * 3 * DB * 2);
  unsigned short* woB_t   = (unsigned short*)allocB((size_t)DB * DB * 2);
  unsigned short* w1B_t   = (unsigned short*)allocB((size_t)DB * DFF * 2);
  unsigned short* w2B_t   = (unsigned short*)allocB((size_t)DFF * DB * 2);
  unsigned short* wk_t    = (unsigned short*)allocB((size_t)AA * DD * DKQ * 2);
  unsigned short* lm_t    = (unsigned short*)allocB((size_t)DB * VV * 2);

  const int tpb = 256;
  auto cast = [&](const float* s, unsigned short* d, long n) {
    k_cast<<<(unsigned)((n / 4 + tpb - 1) / tpb), tpb, 0, stream>>>(s, d, n);
  };
  auto castT = [&](const float* s, unsigned short* d, int rows, int cols) {
    const long n = (long)rows * cols;
    k_cast_t<<<(unsigned)((n + tpb - 1) / tpb), tpb, 0, stream>>>(s, d, rows, cols);
  };
  auto gemmHH = [&](const unsigned short* Am, const unsigned short* Bm, unsigned short* Cm,
                    int M, int N, int K, int act) {
    k_gemm<unsigned short, unsigned short>
        <<<dim3((unsigned)(N / 16), (unsigned)(M / 64), 1), 128, 0, stream>>>(
            Am, Bm, Cm, M, N, K, 0, 0, 0, nullptr, nullptr, act);
  };
  auto gemmHF = [&](const unsigned short* Am, const unsigned short* Bm, float* Cm,
                    int M, int N, int K, long sA, long sB, long sC,
                    const float* bias, const float* resid, int batch) {
    k_gemm<unsigned short, float>
        <<<dim3((unsigned)(N / 16), (unsigned)(M / 64), (unsigned)batch), 128, 0, stream>>>(
            Am, Bm, Cm, M, N, K, sA, sB, sC, bias, resid, 0);
  };
  auto gemmFF = [&](const float* Am, const unsigned short* Bm, float* Cm,
                    int M, int N, int K, long sA, long sB, long sC, int batch) {
    k_gemm<float, float>
        <<<dim3((unsigned)(N / 16), (unsigned)(M / 64), (unsigned)batch), 128, 0, stream>>>(
            Am, Bm, Cm, M, N, K, sA, sB, sC, nullptr, nullptr, 0);
  };
  auto block = [&](float* xb, const float* l1g, const float* l1b,
                   const unsigned short* wqkv_t, const unsigned short* wo_t,
                   const float* l2g, const float* l2b,
                   const unsigned short* w1_t, const unsigned short* w2_t) {
    k_ln<unsigned short><<<BB * TT, 256, 0, stream>>>(xb, lnb_bf, l1g, l1b, DA);
    gemmHH(lnb_bf, wqkv_t, qkv_bf, BB * TT, 3 * DA, DA, 0);
    k_attn<<<dim3(TT / 64, BB * NH), 128, 0, stream>>>(qkv_bf, att_bf);
    gemmHF(att_bf, wo_t, xb, BB * TT, DA, DA, 0, 0, 0, nullptr, xb, 1);       // +resid
    k_ln<unsigned short><<<BB * TT, 256, 0, stream>>>(xb, lnb_bf, l2g, l2b, DA);
    gemmHH(lnb_bf, w1_t, ffn_bf, BB * TT, DFF, DA, 1);                         // gelu
    gemmHF(ffn_bf, w2_t, xb, BB * TT, DA, DFF, 0, 0, 0, nullptr, xb, 1);      // +resid
  };

  // 0) transposed-bf16 weight conversions (amortized over all 2048 token-rows)
  castT(a_wqkv, wqkvA_t, DA, 3 * DA);
  castT(a_wo,   woA_t,   DA, DA);
  castT(a_w1,   w1A_t,   DA, DFF);
  castT(a_w2,   w2A_t,   DFF, DA);
  castT(b_wqkv, wqkvB_t, DB, 3 * DB);
  castT(b_wo,   woB_t,   DB, DB);
  castT(b_w1,   w1B_t,   DB, DFF);
  castT(b_w2,   w2B_t,   DFF, DB);
  for (int a = 0; a < AA; ++a)
    castT(wk + (size_t)a * DD * DKQ, wk_t + (size_t)a * DKQ * DD, DD, DKQ);
  castT(lm_head, lm_t, DB, VV);

  // 1) embed + posenc
  k_embed<<<(BB * TT * DA + tpb - 1) / tpb, tpb, 0, stream>>>(ids, emb, x);
  // 2) transformer block A
  block(x, a_ln1g, a_ln1b, wqkvA_t, woA_t, a_ln2g, a_ln2b, w1A_t, w2A_t);
  // 3) retrieval
  k_rowmean<<<(BB * DA + tpb - 1) / tpb, tpb, 0, stream>>>(x, z);
  k_qvec<<<(BB * AA * DKQ + tpb - 1) / tpb, tpb, 0, stream>>>(z, wq, qv);
  gemmFF(pool, wk_t, keys, NPOOL, DKQ, DD, 0, (long)DKQ * DD, (long)NPOOL * DKQ, AA);
  k_comb<<<(BB * NPOOL + tpb - 1) / tpb, tpb, 0, stream>>>(qv, keys, comb);
  // 4) top-k, alphas, mixed, W assembly (bf16 W^T)
  k_topk<<<BB, 256, 0, stream>>>(comb, tvals, tidx);
  k_alphas<<<1, 32, 0, stream>>>(tvals, lam, warm, alph);
  k_mixed<<<(BB * DD + tpb - 1) / tpb, tpb, 0, stream>>>(alph, tidx, pool, mix);
  k_wassm<<<(BB * DA * DB + tpb - 1) / tpb, tpb, 0, stream>>>(mix, w_base, gamma, Wt_bf);
  // 5) h_mid = LN( bmm(h_A, W) + b_base )
  cast(x, x_bf, (long)BB * TT * DA);
  gemmHF(x_bf, Wt_bf, hmid, TT, DB, DA, (long)TT * DA, (long)DA * DB, (long)TT * DB,
         b_base, nullptr, BB);
  k_ln<float><<<BB * TT, 256, 0, stream>>>(hmid, hmid, mid_g, mid_b, DB);
  // 6) transformer block B
  block(hmid, b_ln1g, b_ln1b, wqkvB_t, woB_t, b_ln2g, b_ln2b, w1B_t, w2B_t);
  // 7) LM head (dominant GEMM, bf16 operands)
  cast(hmid, h_bf, (long)BB * TT * DB);
  gemmHF(h_bf, lm_t, out, BB * TT, VV, DB, 0, 0, 0, nullptr, nullptr, 1);
}
